// SAModule_12060268167708
// MI455X (gfx1250) — compile-verified, matrix operand order checked by reading, not run on hardware
//
#include <hip/hip_runtime.h>

#define N_PTS   16384
#define M_CTR   4096
#define KNBR    64
#define R2      0.0625f
#define CIN     128
#define STRIDE  4          // N/M

#define POS_OFF   (M_CTR*256)
#define BATCH_OFF (POS_OFF + M_CTR*3)
#define LF_OFF    (BATCH_OFF + M_CTR)

typedef _Float16 f16_t;
typedef __attribute__((ext_vector_type(16))) _Float16 v16h;
typedef __attribute__((ext_vector_type(8)))  float    v8f;
typedef __attribute__((ext_vector_type(4)))  unsigned v4u;
typedef __attribute__((ext_vector_type(8)))  int      v8i;
typedef __attribute__((ext_vector_type(4)))  int      v4i;

#define INF_F (__builtin_huge_valf())

// LDS layout for kern_edge (dynamic shared, 252160 B total)
#define L_B1   0            //  65536 B : W1a fragments (f16)
#define L_B2   65536        // 131072 B : W2  fragments (f16)
#define L_XN   196608       //  16384 B : gathered x[nbr] (f16)
#define L_HS   212992       //  32768 B : hidden acts (f16)
#define L_REL  245760       //    768 B
#define L_MSK  246528       //    256 B
#define L_DSTB 246784       //   1024 B
#define L_W1R  247808       //   3072 B
#define L_OUTV 250880       //   1024 B
#define L_NBL  251904       //    256 B
#define SMEM_BYTES 252160

// ---------------------------------------------------------------------------
// Kernel W: convert W1/W2 -> f16 pre-swizzled WMMA B-fragment layout.
// B (32x16 KxN) f16: lane -> N=lane&15 ; VGPR v pair ->
// K = (v&4?16:0) + 2*(v&3) + (lane>=16?8:0) + sub. Fragment = 32B/lane.
// B1 (W1 rows 0..127) and B2 (W2) are CONTIGUOUS in ws (192 KB total).
// ---------------------------------------------------------------------------
__global__ void kern_weights(const float* __restrict__ W1,
                             const float* __restrict__ W2,
                             f16_t* __restrict__ B1,      // [4*16][32][16]
                             f16_t* __restrict__ B2,      // [8*16][32][16]
                             float* __restrict__ relW)    // [3][256]
{
    int g = blockIdx.x * 256 + threadIdx.x;
    if (g < 32768) {
        int f = g >> 9, r = g & 511, l = r >> 4, t = r & 15;
        int kk = f >> 4, j = f & 15, v = t >> 1, sub = t & 1;
        int kl = ((v & 4) << 2) | ((v & 3) << 1) | ((l >> 4) << 3) | sub;
        int n  = j * 16 + (l & 15);
        B1[g] = (f16_t)W1[(kk * 32 + kl) * 256 + n];
    } else if (g < 98304) {
        int q = g - 32768;
        int f = q >> 9, r = q & 511, l = r >> 4, t = r & 15;
        int kk = f >> 4, j = f & 15, v = t >> 1, sub = t & 1;
        int kl = ((v & 4) << 2) | ((v & 3) << 1) | ((l >> 4) << 3) | sub;
        int n  = j * 16 + (l & 15);
        B2[q] = (f16_t)W2[(kk * 32 + kl) * 256 + n];
    } else if (g < 99072) {
        int q = g - 98304;
        relW[q] = W1[(256 + (q >> 8)) * 256 + (q & 255)];
    }
}

// ---------------------------------------------------------------------------
// Kernel P: per-center bias = x_dst @ W1[128:256,:] + b1, + tuple outputs.
// ---------------------------------------------------------------------------
__global__ void kern_bias(const float* __restrict__ x,
                          const float* __restrict__ pos,
                          const int*   __restrict__ batch,
                          const float* __restrict__ lf,
                          const float* __restrict__ W1,
                          const float* __restrict__ b1,
                          float* __restrict__ bias,
                          float* __restrict__ dout)
{
    int m = blockIdx.x, n = threadIdx.x, i0 = m * STRIDE;
    __shared__ float xd[CIN];
    if (n < CIN) xd[n] = x[i0 * CIN + n];
    __syncthreads();
    float acc = b1[n];
    #pragma unroll 8
    for (int c = 0; c < CIN; ++c)
        acc += xd[c] * W1[(CIN + c) * 256 + n];
    bias[m * 256 + n] = acc;
    if (n < 3)              dout[POS_OFF   + m * 3 + n]       = pos[i0 * 3 + n];
    if (n == 3)             dout[BATCH_OFF + m]               = (float)batch[i0];
    if (n >= 4 && n < 13)   dout[LF_OFF    + m * 9 + (n - 4)] = lf[i0 * 9 + (n - 4)];
}

// ---------------------------------------------------------------------------
// Kernel Q: exact K-nearest-within-ball per center (batch-block restricted).
// 64 rounds of packed (f32bits,id) u64 atomicMin in LDS (ds_min_u64).
// ---------------------------------------------------------------------------
__global__ void kern_ballq(const float* __restrict__ pos,
                           int*   __restrict__ nbr,
                           float* __restrict__ maskf)
{
    int m = blockIdx.x, tid = threadIdx.x;
    int i0 = m * STRIDE;
    int base = (i0 >> 11) << 11;
    float px = pos[i0 * 3 + 0], py = pos[i0 * 3 + 1], pz = pos[i0 * 3 + 2];

    float key[8];
    #pragma unroll
    for (int t = 0; t < 8; ++t) {
        int p = base + t * 256 + tid;
        float dx = pos[p * 3 + 0] - px;
        float dy = pos[p * 3 + 1] - py;
        float dz = pos[p * 3 + 2] - pz;
        float d2 = dx * dx + dy * dy + dz * dz;
        key[t] = (d2 <= R2) ? d2 : INF_F;
    }

    __shared__ unsigned long long best;
    for (int r = 0; r < KNBR; ++r) {
        if (tid == 0) best = ~0ull;
        __syncthreads();
        float km = key[0]; int tm = 0;
        #pragma unroll
        for (int t = 1; t < 8; ++t)
            if (key[t] < km) { km = key[t]; tm = t; }
        unsigned long long pk =
            ((unsigned long long)__float_as_uint(km) << 32) |
            (unsigned)(tm * 256 + tid);
        atomicMin(&best, pk);
        __syncthreads();
        unsigned long long b = best;
        unsigned id = (unsigned)(b & 0xffffffffu);
        unsigned kb = (unsigned)(b >> 32);
        if ((id & 255u) == (unsigned)tid) key[id >> 8] = INF_F;
        if (tid == 0) {
            bool valid = kb < 0x7f800000u;
            nbr[m * KNBR + r]   = valid ? (base + (int)id) : i0;
            maskf[m * KNBR + r] = valid ? 0.0f : -INF_F;
        }
        __syncthreads();
    }
}

// ---------------------------------------------------------------------------
// Kernel E: fused EdgeConv MLP. Weights staged ONCE per block in LDS (via
// TDM tensor_load_to_lds when available), WMMA B-frags come from LDS.
// ---------------------------------------------------------------------------
union AFrag { v16h h; unsigned u[8]; };

__global__ void __launch_bounds__(256)
kern_edge(const float* __restrict__ x,
          const float* __restrict__ pos,
          const float* __restrict__ lf,
          const f16_t* __restrict__ Bw,    // B1|B2 contiguous, 196608 B
          const float* __restrict__ relW,
          const float* __restrict__ bias,
          const int*   __restrict__ nbrG,
          const float* __restrict__ maskG,
          const float* __restrict__ b2,
          float* __restrict__ dout)
{
    extern __shared__ __align__(16) char smem[];
    f16_t* xn   = (f16_t*)(smem + L_XN);
    f16_t* Hs   = (f16_t*)(smem + L_HS);
    float* relb = (float*)(smem + L_REL);
    float* mskb = (float*)(smem + L_MSK);
    float* dstb = (float*)(smem + L_DSTB);
    float* w1r  = (float*)(smem + L_W1R);
    float* outv = (float*)(smem + L_OUTV);
    int*   nbl  = (int*)  (smem + L_NBL);
    const v16h* B1f = (const v16h*)(smem + L_B1);
    const v16h* B2f = (const v16h*)(smem + L_B2);

    const int tid  = threadIdx.x;
    const int w    = tid >> 5;
    const int lane = tid & 31;
    const int rt   = w >> 1;
    const int cb   = (w & 1) * 8;
    const int mrow = lane & 15;
    const int hi8  = (lane >> 4) << 3;

    // ---- stage weights into LDS (once per block) ----
#if __has_builtin(__builtin_amdgcn_tensor_load_to_lds) && __has_builtin(__builtin_amdgcn_s_wait_tensorcnt)
    if (w == 0) {
        unsigned lds_base = (unsigned)(unsigned long long)(smem + L_B1);
        unsigned long long ga = (unsigned long long)Bw;
        // D# group0: count=1, lds_addr, global_addr[56:0], type=2
        v4u g0 = { 1u, lds_base,
                   (unsigned)(ga & 0xffffffffu),
                   ((unsigned)((ga >> 32) & 0x01ffffffu)) | (2u << 30) };
        // D# group1: data_size=8B; tensor_dim0=tile_dim0=stride0=24576 (x8B
        // = 196608 B); tensor_dim1=tile_dim1=1; stride1=24576.
        v8i g1 = { (int)0x00030000, (int)0x60000000, (int)0x00010000,
                   (int)0x60000000, (int)0x00000001, (int)0x00006000,
                   (int)0x60000000, 0 };
        v4i g2 = { 0, 0, 0, 0 };
        v4i g3 = { 0, 0, 0, 0 };
#if __clang_major__ >= 23
        v8i g4 = { 0, 0, 0, 0, 0, 0, 0, 0 };
        __builtin_amdgcn_tensor_load_to_lds(g0, g1, g2, g3, g4, 0);
#else
        __builtin_amdgcn_tensor_load_to_lds(g0, g1, g2, g3, 0);
#endif
        __builtin_amdgcn_s_wait_tensorcnt(0);
    }
#else
    for (int i = tid; i < 12288; i += 256)
        ((uint4*)smem)[i] = ((const uint4*)Bw)[i];
#endif
    for (int i = tid; i < 768; i += 256) w1r[i] = relW[i];
    __syncthreads();

    const v8f zero = {0.f, 0.f, 0.f, 0.f, 0.f, 0.f, 0.f, 0.f};

    for (int m = blockIdx.x; m < M_CTR; m += gridDim.x) {
        __builtin_prefetch(&bias[(m + gridDim.x < M_CTR ? m + gridDim.x : m) * 256], 0, 1);

        // per-center metadata + rel_loc = lframes_dst @ (pos[nbr]-pos_dst)
        if (tid < KNBR) {
            int i0 = m * STRIDE;
            int nb = nbrG[m * KNBR + tid];
            nbl[tid]  = nb;
            mskb[tid] = maskG[m * KNBR + tid];
            float dx = pos[nb * 3 + 0] - pos[i0 * 3 + 0];
            float dy = pos[nb * 3 + 1] - pos[i0 * 3 + 1];
            float dz = pos[nb * 3 + 2] - pos[i0 * 3 + 2];
            #pragma unroll
            for (int i = 0; i < 3; ++i)
                relb[tid * 3 + i] = lf[i0 * 9 + i * 3 + 0] * dx
                                  + lf[i0 * 9 + i * 3 + 1] * dy
                                  + lf[i0 * 9 + i * 3 + 2] * dz;
        }
        dstb[tid] = bias[m * 256 + tid];
        __syncthreads();

        // gather neighbor features -> f16 LDS
        for (int e = tid; e < 64 * 128; e += 256) {
            int row = e >> 7, col = e & 127;
            xn[e] = (f16_t)x[nbl[row] * CIN + col];
        }
        __syncthreads();

        // -------- GEMM1: 64x256 = xn(64x128) @ W1a --------
        v8f acc[8];
        #pragma unroll
        for (int c = 0; c < 8; ++c) acc[c] = zero;
        #pragma unroll
        for (int kk = 0; kk < 4; ++kk) {
            AFrag a;
            const char* ab = (const char*)xn + 2 * ((rt * 16 + mrow) * 128 + kk * 32);
            #pragma unroll
            for (int v = 0; v < 8; ++v) {
                int ko = ((v & 4) << 2) | ((v & 3) << 1) | hi8;
                a.u[v] = *(const unsigned*)(ab + 2 * ko);
            }
            #pragma unroll
            for (int c = 0; c < 8; ++c) {
                v16h bfr = B1f[(kk * 16 + cb + c) * 32 + lane];
                acc[c] = __builtin_amdgcn_wmma_f32_16x16x32_f16(
                    false, a.h, false, bfr, (short)0, acc[c], false, false);
            }
        }
        // epilogue 1: + bias + rel.W1rel, relu, -> f16 H
        #pragma unroll
        for (int c = 0; c < 8; ++c) {
            #pragma unroll
            for (int v = 0; v < 8; ++v) {
                int ke = rt * 16 + v + hi8;
                int n  = (cb + c) * 16 + mrow;
                float h = acc[c][v] + dstb[n]
                        + relb[ke * 3 + 0] * w1r[n]
                        + relb[ke * 3 + 1] * w1r[256 + n]
                        + relb[ke * 3 + 2] * w1r[512 + n];
                h = fmaxf(h, 0.0f);
                Hs[ke * 256 + n] = (f16_t)h;
            }
        }
        __syncthreads();
        outv[tid] = -INF_F;
        __syncthreads();

        // -------- GEMM2: 64x256 = H(64x256) @ W2 --------
        #pragma unroll
        for (int c = 0; c < 8; ++c) acc[c] = zero;
        #pragma unroll
        for (int kk = 0; kk < 8; ++kk) {
            AFrag a;
            const char* ab = (const char*)Hs + 2 * ((rt * 16 + mrow) * 256 + kk * 32);
            #pragma unroll
            for (int v = 0; v < 8; ++v) {
                int ko = ((v & 4) << 2) | ((v & 3) << 1) | hi8;
                a.u[v] = *(const unsigned*)(ab + 2 * ko);
            }
            #pragma unroll
            for (int c = 0; c < 8; ++c) {
                v16h bfr = B2f[(kk * 16 + cb + c) * 32 + lane];
                acc[c] = __builtin_amdgcn_wmma_f32_16x16x32_f16(
                    false, a.h, false, bfr, (short)0, acc[c], false, false);
            }
        }
        // epilogue 2: masked max over edge rows -> LDS fmax atomic
        #pragma unroll
        for (int c = 0; c < 8; ++c) {
            #pragma unroll
            for (int v = 0; v < 8; ++v) {
                int ke = rt * 16 + v + hi8;
                int n  = (cb + c) * 16 + mrow;
                float mv = acc[c][v] + mskb[ke];   // -inf for invalid edges
                __hip_atomic_fetch_max(&outv[n], mv, __ATOMIC_RELAXED,
                                       __HIP_MEMORY_SCOPE_WORKGROUP);
            }
        }
        __syncthreads();

        float r = outv[tid];
        dout[m * 256 + tid] = (r == -INF_F) ? 0.0f : r + b2[tid];
        __syncthreads();   // LDS reuse next center
    }
}

// ---------------------------------------------------------------------------
extern "C" void kernel_launch(void* const* d_in, const int* in_sizes, int n_in,
                              void* d_out, int out_size, void* d_ws, size_t ws_size,
                              hipStream_t stream) {
    const float* x     = (const float*)d_in[0];
    const float* pos   = (const float*)d_in[1];
    const int*   batch = (const int*)  d_in[2];
    const float* lf    = (const float*)d_in[3];
    const float* W1    = (const float*)d_in[4];
    const float* b1    = (const float*)d_in[5];
    const float* W2    = (const float*)d_in[6];
    const float* b2    = (const float*)d_in[7];
    float* dout = (float*)d_out;

    char* ws = (char*)d_ws;
    f16_t* wsB1  = (f16_t*)(ws);                                   //  64 KB
    f16_t* wsB2  = (f16_t*)(ws + 65536);                           // 128 KB
    float* wsRel = (float*)(ws + 196608);                          //   3 KB
    float* wsBia = (float*)(ws + 199680);                          //   4 MB
    int*   wsNbr = (int*)  (ws + 199680 + 4194304);                //   1 MB
    float* wsMsk = (float*)(ws + 199680 + 4194304 + 1048576);      //   1 MB

    (void)hipFuncSetAttribute((const void*)kern_edge,
                              hipFuncAttributeMaxDynamicSharedMemorySize,
                              SMEM_BYTES);

    kern_weights<<<(99072 + 255) / 256, 256, 0, stream>>>(W1, W2, wsB1, wsB2, wsRel);
    kern_bias<<<M_CTR, 256, 0, stream>>>(x, pos, batch, lf, W1, b1, wsBia, dout);
    kern_ballq<<<M_CTR, 256, 0, stream>>>(pos, wsNbr, wsMsk);
    kern_edge<<<512, 256, SMEM_BYTES, stream>>>(x, pos, lf, wsB1, wsRel, wsBia,
                                                wsNbr, wsMsk, b2, dout);
}
